// QuantPNet_64201171141240
// MI455X (gfx1250) — 8-bit matrix operand order repaired and checked by reading, compiled, not run on hardware
//
#include <hip/hip_runtime.h>
#include <stdint.h>

typedef __attribute__((ext_vector_type(8))) int v8i;
union V8 { v8i v; int i[8]; unsigned u[8]; };

static constexpr int NB = 32;
static constexpr int H0 = 384, W0 = 384, C0 = 3;
static constexpr int C1 = 10, H1 = 382, W1 = 382;
static constexpr int HP = 191, WP = 191;
static constexpr int C2 = 16, H2 = 189, W2 = 189;
static constexpr int C3 = 32, H3 = 187, W3 = 187;

static constexpr long SZ_X  = (long)NB * C0 * H0 * W0;
static constexpr long SZ_R1 = (long)NB * C1 * H1 * W1;
static constexpr long SZ_P  = (long)NB * C1 * HP * WP;
static constexpr long SZ_C2 = (long)NB * C2 * H2 * W2;
static constexpr long SZ_C3 = (long)NB * C3 * H3 * W3;
static constexpr long SZ_HB = (long)NB * 4 * H3 * W3;
static constexpr long SZ_HA = (long)NB * 2 * H3 * W3;

static constexpr int M2 = NB * H2 * W2;
static constexpr int M3 = NB * H3 * W3;

static constexpr size_t OFF_SCALES = 0;
static constexpr size_t OFF_BP2  = 256;
static constexpr size_t OFF_BP3  = OFF_BP2 + 2048;
static constexpr size_t OFF_W1Q  = OFF_BP3 + 6144;
static constexpr size_t OFF_W41Q = OFF_W1Q + 512;
static constexpr size_t OFF_W42Q = OFF_W41Q + 64;
static constexpr size_t OFF_QX   = 16384;
static constexpr size_t OFF_Q1   = OFF_QX + (size_t)SZ_X;
static constexpr size_t OFF_QP   = OFF_Q1 + (size_t)SZ_R1;
static constexpr size_t OFF_Q2   = OFF_QX;
static constexpr size_t OFF_Q3   = OFF_QP + (size_t)SZ_P;
static constexpr size_t OFF_F    = ((OFF_Q3 + (size_t)SZ_C3 + 255) / 256) * 256;
static constexpr size_t WS_NEED  = OFF_F + (size_t)SZ_R1 * 4;

__device__ __forceinline__ float wave_max32(float m) {
#pragma unroll
  for (int off = 16; off; off >>= 1) m = fmaxf(m, __shfl_xor(m, off, 32));
  return m;
}
__device__ __forceinline__ float rd_scale(const float* s, int i, float hi) {
  return fmaxf(s[i], 1e-8f) / hi;
}

__global__ void k_zero_scales(unsigned* s) {
  if (threadIdx.x < 16) s[threadIdx.x] = 0u;
}

__global__ void k_absmax(const float* __restrict__ p, long n, unsigned* dst) {
  float m = 0.f;
  for (long i = (long)blockIdx.x * blockDim.x + threadIdx.x; i < n;
       i += (long)gridDim.x * blockDim.x)
    m = fmaxf(m, fabsf(p[i]));
  m = wave_max32(m);
  if ((threadIdx.x & 31) == 0) atomicMax(dst, __float_as_uint(m));
}

__global__ void k_quant_x(const float* __restrict__ x, const float* __restrict__ sc,
                          int8_t* __restrict__ q, long n) {
  long i = (long)blockIdx.x * blockDim.x + threadIdx.x;
  if (i >= n) return;
  float s = rd_scale(sc, 0, 7.0f);
  float c = fminf(7.f, fmaxf(-8.f, rintf(x[i] / s)));
  q[i] = (int8_t)(int)c;
}

__global__ void k_quant_u4(const float* __restrict__ v, const float* __restrict__ sc,
                           int si, uint8_t* __restrict__ q, long n) {
  long i = (long)blockIdx.x * blockDim.x + threadIdx.x;
  if (i >= n) return;
  float s = rd_scale(sc, si, 15.0f);
  float c = fminf(15.f, fmaxf(0.f, rintf(v[i] / s)));
  q[i] = (uint8_t)(int)c;
}

__global__ void k_quant_w(const float* __restrict__ w, const float* __restrict__ sc,
                          int si, int8_t* __restrict__ q, int n) {
  int i = blockIdx.x * blockDim.x + threadIdx.x;
  if (i >= n) return;
  float s = rd_scale(sc, si, 7.0f);
  float c = fminf(7.f, fmaxf(-7.f, rintf(w[i] / s)));
  q[i] = (int8_t)(int)c;
}

__global__ void k_pack_b(const float* __restrict__ w, const float* __restrict__ sc,
                         int si, unsigned* __restrict__ dst,
                         int cout, int cin, int nchunk) {
  int t = blockIdx.x * blockDim.x + threadIdx.x;
  int total = (cout / 16) * nchunk * 256;
  if (t >= total) return;
  int j = t & 7, lane = (t >> 3) & 31, rest = t >> 8;
  int chunk = rest % nchunk, nt = rest / nchunk;
  int lanehi = lane >> 4;
  int oc = nt * 16 + (lane & 15);
  int ktot = cin * 9;
  float s = rd_scale(sc, si, 7.0f);
  int koff = (j < 4 ? j * 4 : 32 + (j - 4) * 4) + lanehi * 16;
  unsigned dw = 0;
#pragma unroll
  for (int b = 0; b < 4; ++b) {
    int K = chunk * 64 + koff + b;
    int vv = 0;
    if (K < ktot) {
      int c = K / 9, tap = K % 9, ky = tap / 3, kx = tap % 3;
      float wf = w[((oc * cin + c) * 3 + ky) * 3 + kx];
      vv = (int)fminf(7.f, fmaxf(-7.f, rintf(wf / s)));
    }
    dw |= ((unsigned)(vv & 0xFF)) << (8 * b);
  }
  dst[t] = dw;
}

__global__ __launch_bounds__(256)
void k_conv1(const int8_t* __restrict__ qx, const int8_t* __restrict__ wq,
             const float* __restrict__ b1, const float* __restrict__ sc,
             float* __restrict__ out, unsigned* __restrict__ amax) {
  long i = (long)blockIdx.x * blockDim.x + threadIdx.x;
  float lmax = 0.f;
  if (i < SZ_R1) {
    int ox = (int)(i % W1);
    long r = i / W1;
    int oy = (int)(r % H1); r /= H1;
    int oc = (int)(r % C1);
    int n = (int)(r / C1);
    int acc = 0;
#pragma unroll
    for (int c = 0; c < C0; ++c)
#pragma unroll
      for (int ky = 0; ky < 3; ++ky)
#pragma unroll
        for (int kx = 0; kx < 3; ++kx)
          acc += (int)qx[(((long)(n * C0 + c) * H0) + oy + ky) * W0 + ox + kx] *
                 (int)wq[((oc * C0 + c) * 3 + ky) * 3 + kx];
    float bs = rd_scale(sc, 0, 7.0f) * rd_scale(sc, 1, 7.0f);
    float y = (float)acc * bs + rintf(b1[oc] / bs) * bs;
    y = fmaxf(y, 0.f);
    out[i] = y;
    lmax = y;
  }
  lmax = wave_max32(lmax);
  if ((threadIdx.x & 31) == 0) atomicMax(amax, __float_as_uint(lmax));
}

__global__ void k_pool(const uint8_t* __restrict__ q, uint8_t* __restrict__ o) {
  long i = (long)blockIdx.x * blockDim.x + threadIdx.x;
  if (i >= SZ_P) return;
  int px = (int)(i % WP);
  long r = i / WP;
  int py = (int)(r % HP); r /= HP;
  long nc = r;
  long base = (nc * H1 + 2 * py) * W1 + 2 * px;
  int m = q[base];
  if (2 * px + 1 < W1) m = max(m, (int)q[base + 1]);
  if (2 * py + 1 < H1) {
    m = max(m, (int)q[base + W1]);
    if (2 * px + 1 < W1) m = max(m, (int)q[base + W1 + 1]);
  }
  o[i] = (uint8_t)m;
}

template <int CIN, int NCHUNK, int COUT>
__global__ __launch_bounds__(256)
void k_conv_wmma(const uint8_t* __restrict__ qin, const unsigned* __restrict__ bpack,
                 const float* __restrict__ bias, const float* __restrict__ sc,
                 int si_in, int si_w, float* __restrict__ out,
                 unsigned* __restrict__ amax,
                 int Hin, int Win, int Hout, int Wout, int Mtotal) {
  constexpr int KPAD = NCHUNK * 64;
  constexpr int KTOT = CIN * 9;
  __shared__ uint8_t tileA[128 * KPAD];

  const int m0 = blockIdx.x * 128;
  const int hw = Hout * Wout;

  for (int t = threadIdx.x; t < 128 * KPAD; t += 256) {
    int mloc = t / KPAD, k = t - mloc * KPAD;
    uint8_t v = 0;
    int m = m0 + mloc;
    if (m < Mtotal && k < KTOT) {
      int c = k / 9, tap = k - 9 * c, ky = tap / 3, kx = tap - 3 * ky;
      int n = m / hw, r = m - n * hw, oy = r / Wout, ox = r - oy * Wout;
      const uint8_t* src =
          qin + (((long)(n * CIN + c) * Hin) + oy + ky) * Win + ox + kx;
      v = *src;
      __builtin_prefetch(src + Win, 0, 1);
    }
    tileA[t] = v;
  }
  __syncthreads();

  const int wv = threadIdx.x >> 5, ln = threadIdx.x & 31;
  const int lanehi = ln >> 4, col = ln & 15, arow = ln & 15;
  const int rowbase = wv * 16;
  const float s_in = rd_scale(sc, si_in, 15.0f);
  const float s_w = rd_scale(sc, si_w, 7.0f);
  const float bs = s_in * s_w;
  float lmax = 0.f;

#pragma unroll
  for (int nt = 0; nt < COUT / 16; ++nt) {
    V8 acc;
#pragma unroll
    for (int j = 0; j < 8; ++j) acc.i[j] = 0;

#pragma unroll
    for (int ch = 0; ch < NCHUNK; ++ch) {
      V8 a, b;
#pragma unroll
      for (int j = 0; j < 8; ++j) {
        int koff = (j & 3) * 4 + ((j & 4) ? 32 : 0) + lanehi * 16;
        a.i[j] = *(const int*)&tileA[(rowbase + arow) * KPAD + ch * 64 + koff];
      }
      const unsigned* bp = bpack + (((nt * NCHUNK + ch) * 32) + ln) * 8;
#pragma unroll
      for (int j = 0; j < 8; ++j) b.u[j] = bp[j];
      acc.v = __builtin_amdgcn_wmma_i32_16x16x64_iu8(false, a.v, true, b.v,
                                                     acc.v, false, false);
    }

    int oc = nt * 16 + col;
    float bq = rintf(bias[oc] / bs) * bs;
#pragma unroll
    for (int r2 = 0; r2 < 8; ++r2) {
      int m = m0 + rowbase + lanehi * 8 + r2;
      if (m < Mtotal) {
        float y = (float)acc.i[r2] * bs + bq;
        y = fmaxf(y, 0.f);
        lmax = fmaxf(lmax, y);
        int n = m / hw, rr = m - n * hw, oy = rr / Wout, ox = rr - oy * Wout;
        out[(((long)(n * COUT + oc) * Hout) + oy) * Wout + ox] = y;
      }
    }
  }
  lmax = wave_max32(lmax);
  if (ln == 0) atomicMax(amax, __float_as_uint(lmax));
}

__global__ __launch_bounds__(256)
void k_heads(const uint8_t* __restrict__ q3, const int8_t* __restrict__ w41q,
             const int8_t* __restrict__ w42q, const float* __restrict__ b41,
             const float* __restrict__ b42, const float* __restrict__ sc,
             float* __restrict__ outp) {
  long pix = (long)blockIdx.x * blockDim.x + threadIdx.x;
  const long hw = (long)H3 * W3;
  if (pix >= (long)NB * hw) return;
  int n = (int)(pix / hw);
  long r = pix - (long)n * hw;

  int u[C3];
  const uint8_t* base = q3 + ((long)n * C3) * hw + r;
#pragma unroll
  for (int c = 0; c < C3; ++c) u[c] = base[(long)c * hw];

  float s3 = rd_scale(sc, 8, 15.0f);
  float s41 = rd_scale(sc, 4, 7.0f), s42 = rd_scale(sc, 5, 7.0f);
  float bsA = s3 * s41, bsB = s3 * s42;

  float a[2];
#pragma unroll
  for (int k = 0; k < 2; ++k) {
    int acc = 0;
#pragma unroll
    for (int c = 0; c < C3; ++c) acc += u[c] * (int)w41q[k * C3 + c];
    a[k] = (float)acc * bsA + rintf(b41[k] / bsA) * bsA;
  }
  float mx = fmaxf(a[0], a[1]);
  float e0 = __expf(a[0] - mx), e1 = __expf(a[1] - mx);
  float inv = 1.f / (e0 + e1);

  float* outb = outp;
  float* outa = outp + SZ_HB;
#pragma unroll
  for (int k = 0; k < 4; ++k) {
    int acc = 0;
#pragma unroll
    for (int c = 0; c < C3; ++c) acc += u[c] * (int)w42q[k * C3 + c];
    outb[((long)(n * 4 + k)) * hw + r] =
        (float)acc * bsB + rintf(b42[k] / bsB) * bsB;
  }
  outa[((long)(n * 2 + 0)) * hw + r] = e0 * inv;
  outa[((long)(n * 2 + 1)) * hw + r] = e1 * inv;
}

extern "C" void kernel_launch(void* const* d_in, const int* in_sizes, int n_in,
                              void* d_out, int out_size, void* d_ws, size_t ws_size,
                              hipStream_t stream) {
  (void)in_sizes; (void)n_in; (void)out_size;
  if (ws_size < WS_NEED) return;

  const float* x   = (const float*)d_in[0];
  const float* w1  = (const float*)d_in[1];
  const float* b1  = (const float*)d_in[2];
  const float* w2  = (const float*)d_in[3];
  const float* b2  = (const float*)d_in[4];
  const float* w3  = (const float*)d_in[5];
  const float* b3  = (const float*)d_in[6];
  const float* w41 = (const float*)d_in[7];
  const float* b41 = (const float*)d_in[8];
  const float* w42 = (const float*)d_in[9];
  const float* b42 = (const float*)d_in[10];

  char* ws = (char*)d_ws;
  unsigned* scU = (unsigned*)(ws + OFF_SCALES);
  float*    scF = (float*)(ws + OFF_SCALES);
  unsigned* bp2 = (unsigned*)(ws + OFF_BP2);
  unsigned* bp3 = (unsigned*)(ws + OFF_BP3);
  int8_t*   w1q = (int8_t*)(ws + OFF_W1Q);
  int8_t*  w41q = (int8_t*)(ws + OFF_W41Q);
  int8_t*  w42q = (int8_t*)(ws + OFF_W42Q);
  int8_t*    qx = (int8_t*)(ws + OFF_QX);
  uint8_t*   q1 = (uint8_t*)(ws + OFF_Q1);
  uint8_t*   qp = (uint8_t*)(ws + OFF_QP);
  uint8_t*   q2 = (uint8_t*)(ws + OFF_Q2);
  uint8_t*   q3 = (uint8_t*)(ws + OFF_Q3);
  float*      F = (float*)(ws + OFF_F);
  float*    out = (float*)d_out;

  k_zero_scales<<<1, 32, 0, stream>>>(scU);
  k_absmax<<<2048, 256, 0, stream>>>(x, SZ_X, scU + 0);
  k_absmax<<<1, 256, 0, stream>>>(w1, 270, scU + 1);
  k_absmax<<<2, 256, 0, stream>>>(w2, 1440, scU + 2);
  k_absmax<<<8, 256, 0, stream>>>(w3, 4608, scU + 3);
  k_absmax<<<1, 64, 0, stream>>>(w41, 64, scU + 4);
  k_absmax<<<1, 128, 0, stream>>>(w42, 128, scU + 5);

  k_quant_x<<<(int)((SZ_X + 255) / 256), 256, 0, stream>>>(x, scF, qx, SZ_X);
  k_quant_w<<<2, 256, 0, stream>>>(w1, scF, 1, w1q, 270);
  k_quant_w<<<1, 64, 0, stream>>>(w41, scF, 4, w41q, 64);
  k_quant_w<<<1, 128, 0, stream>>>(w42, scF, 5, w42q, 128);
  k_pack_b<<<2, 256, 0, stream>>>(w2, scF, 2, bp2, C2, C1, 2);
  k_pack_b<<<6, 256, 0, stream>>>(w3, scF, 3, bp3, C3, C2, 3);

  k_conv1<<<(int)((SZ_R1 + 255) / 256), 256, 0, stream>>>(qx, w1q, b1, scF, F, scU + 6);
  k_quant_u4<<<(int)((SZ_R1 + 255) / 256), 256, 0, stream>>>(F, scF, 6, q1, SZ_R1);
  k_pool<<<(int)((SZ_P + 255) / 256), 256, 0, stream>>>(q1, qp);

  k_conv_wmma<C1, 2, C2><<<(M2 + 127) / 128, 256, 0, stream>>>(
      qp, bp2, b2, scF, 6, 2, F, scU + 7, HP, WP, H2, W2, M2);
  k_quant_u4<<<(int)((SZ_C2 + 255) / 256), 256, 0, stream>>>(F, scF, 7, q2, SZ_C2);

  k_conv_wmma<C2, 3, C3><<<(M3 + 127) / 128, 256, 0, stream>>>(
      q2, bp3, b3, scF, 7, 3, F, scU + 8, H2, W2, H3, W3, M3);
  k_quant_u4<<<(int)((SZ_C3 + 255) / 256), 256, 0, stream>>>(F, scF, 8, q3, SZ_C3);

  k_heads<<<(int)(((long)NB * H3 * W3 + 255) / 256), 256, 0, stream>>>(
      q3, w41q, w42q, b41, b42, scF, out);
}
